// GCL_21560735826060
// MI455X (gfx1250) — matrix operand order repaired and checked by reading, not compiled, not run on hardware
//
#include <hip/hip_runtime.h>
#include <hip/hip_bf16.h>

// ---------------------------------------------------------------------------
// EGNN layer for MI455X (gfx1250, wave32, WMMA).
// Edge MLP + scatter-sum + node MLP, bf16 WMMA with f32 accumulation.
// Memory-bound workload (~650MB effective traffic at 23.3 TB/s); bf16 gather
// table halves the dominant h[row]/h[col] gather stream.
// ---------------------------------------------------------------------------

typedef __bf16 bf16_t;
typedef __attribute__((ext_vector_type(16))) __bf16 v16bf;
typedef __attribute__((ext_vector_type(8)))  float  v8f;

#define N_NODES 50000
#define N_EDGES 600000
#define DD      128
#define WAVES   4          // waves per block, one 16-row tile per wave

__device__ __forceinline__ float silu_f(float x) {
    return x / (1.0f + __expf(-x));
}

// ---------------------------------------------------------------------------
// Weight swizzle: row-major f32 W[K][128] -> bf16 B-fragments.
// Fragment layout (16-bit B, 32x16 tile, wave32):
//   lane L holds column N = t*16 + (L&15); element i holds K = s*32 + 8*(L>>4) + i (i<8), s*32 + 16 + 8*(L>>4) + (i-8) (i>=8)
// Stored as [s][t][lane][i] contiguous -> each lane reads 32B contiguously.
// ---------------------------------------------------------------------------
__global__ void prep_weights(const float* __restrict__ ew1, const float* __restrict__ ew2,
                             const float* __restrict__ nw1, const float* __restrict__ nw2,
                             bf16_t* __restrict__ wb1, bf16_t* __restrict__ wb2,
                             bf16_t* __restrict__ nwb1, bf16_t* __restrict__ nwb2) {
    int id = blockIdx.x * blockDim.x + threadIdx.x;
    if (id >= 8 * 8 * 32 * 16) return;
    int i    = id & 15;
    int lane = (id >> 4) & 31;
    int t    = (id >> 9) & 7;
    int s    = id >> 12;
    int half = lane >> 4;
    int n    = t * 16 + (lane & 15);
    int k    = s * 32 + (i < 8 ? 8 * half + i : 16 + 8 * half + (i - 8));
    wb1[id]  = (bf16_t)ew1[k * DD + n];
    nwb1[id] = (bf16_t)nw1[k * DD + n];
    if (s < 4) {                       // 128-K weights: ids [0, 16384)
        wb2[id]  = (bf16_t)ew2[k * DD + n];
        nwb2[id] = (bf16_t)nw2[k * DD + n];
    }
}

// h (f32) -> bf16 gather table; zero agg scratch (must happen every launch).
__global__ void conv_h(const float* __restrict__ h, bf16_t* __restrict__ hb,
                       float* __restrict__ agg) {
    int id = blockIdx.x * blockDim.x + threadIdx.x;
    if (id >= N_NODES * DD) return;
    hb[id]  = (bf16_t)h[id];
    agg[id] = 0.0f;
}

// A-fragment loader: 16-bit A 16x32 layout = two contiguous 16B chunks in LDS.
//   elements 0..7  : K = s*32 + 8*half + (0..7)
//   elements 8..15 : K = s*32 + 16 + 8*half + (0..7)
__device__ __forceinline__ v16bf load_a_frag(const bf16_t* rowBase, int s, int half) {
    union { v16bf v; uint4 q[2]; } u;
    u.q[0] = *(const uint4*)(rowBase + s * 32 + 8 * half);
    u.q[1] = *(const uint4*)(rowBase + s * 32 + 16 + 8 * half);
    return u.v;
}

// One MLP layer on a 16-row tile held in LDS (cols [0, ksteps*32)):
// acc[t] += A(16 x 32k) * B(32k x 16t). All A/B loads of a K-step are issued
// first, then a scheduling fence, then the 8 independent WMMAs back-to-back.
// The fence stops the backend from sinking B loads onto a single register
// pair (which would serialize every WMMA behind a load round-trip).
template <int KSTEPS>
__device__ __forceinline__ void tile_mlp(const bf16_t* aRow, const v16bf* B,
                                         int lane, int half, v8f acc[8]) {
#pragma unroll
    for (int s = 0; s < KSTEPS; ++s) {
        v16bf a = load_a_frag(aRow, s, half);
        v16bf bf[8];
#pragma unroll
        for (int t = 0; t < 8; ++t) bf[t] = B[(s * 8 + t) * 32 + lane];
        __builtin_amdgcn_sched_barrier(0);   // keep all 8 B frags live
#pragma unroll
        for (int t = 0; t < 8; ++t) {
            acc[t] = __builtin_amdgcn_wmma_f32_16x16x32_bf16(
                false, a, false, bf[t], (short)0, acc[t], false, false);
        }
        __builtin_amdgcn_sched_barrier(0);
    }
}

// ---------------------------------------------------------------------------
// Edge kernel: one wave computes mij for 16 edges (16x128 tile).
// ---------------------------------------------------------------------------
__global__ __launch_bounds__(32 * WAVES) void edge_kernel(
    const bf16_t* __restrict__ hb, const long long* __restrict__ ei,
    const float* __restrict__ eb1, const float* __restrict__ eb2,
    const bf16_t* __restrict__ wb1, const bf16_t* __restrict__ wb2,
    float* __restrict__ mij, float* __restrict__ agg) {

    __shared__ __align__(16) bf16_t at[WAVES][16][2 * DD];

    const int lane = threadIdx.x & 31;
    const int wave = threadIdx.x >> 5;
    const int tile = blockIdx.x * WAVES + wave;
    if (tile >= N_EDGES / 16) return;            // wave-uniform

    const long long* rowI = ei;
    const long long* colI = ei + N_EDGES;
    const long long  e0   = (long long)tile * 16;

    // Stage 16 edges x 256 bf16 (concat of both endpoints) via b128 copies.
    for (int idx = lane; idx < 512; idx += 32) {
        int m = idx >> 5;
        int p = (idx >> 4) & 1;
        int v = idx & 15;
        long long node = p ? colI[e0 + m] : rowI[e0 + m];
        ((uint4*)&at[wave][m][p * DD])[v] = ((const uint4*)(hb + node * DD))[v];
    }
    asm volatile("s_wait_dscnt 0" ::: "memory");

    const int half = lane >> 4;
    const int m16  = lane & 15;
    const bf16_t* aRow = &at[wave][m16][0];

    // ---- layer 1: [16,256] x [256,128] ----
    v8f acc[8];
#pragma unroll
    for (int t = 0; t < 8; ++t) {
        float b = eb1[t * 16 + m16];
#pragma unroll
        for (int r = 0; r < 8; ++r) acc[t][r] = b;
    }
    tile_mlp<8>(aRow, (const v16bf*)wb1, lane, half, acc);

    // SiLU -> bf16 layer-2 A tile (cols 0..127 of LDS tile). D layout: row = r + 8*half.
#pragma unroll
    for (int t = 0; t < 8; ++t)
#pragma unroll
        for (int r = 0; r < 8; ++r)
            at[wave][r + 8 * half][t * 16 + m16] = (bf16_t)silu_f(acc[t][r]);
    asm volatile("s_wait_dscnt 0" ::: "memory");

    // ---- layer 2: [16,128] x [128,128] ----
    v8f acc2[8];
#pragma unroll
    for (int t = 0; t < 8; ++t) {
        float b = eb2[t * 16 + m16];
#pragma unroll
        for (int r = 0; r < 8; ++r) acc2[t][r] = b;
    }
    tile_mlp<4>(aRow, (const v16bf*)wb2, lane, half, acc2);

    // SiLU, write mij (f32) and scatter-sum into agg[row].
#pragma unroll
    for (int r = 0; r < 8; ++r) {
        int m = r + 8 * half;
        long long e  = e0 + m;
        long long rn = rowI[e];
#pragma unroll
        for (int t = 0; t < 8; ++t) {
            float x = silu_f(acc2[t][r]);
            int n = t * 16 + m16;
            mij[e * DD + n] = x;
            atomicAdd(&agg[rn * DD + n], x);
        }
    }
}

// ---------------------------------------------------------------------------
// Node kernel: one wave computes h_out for 16 nodes (16x128 tile).
// ---------------------------------------------------------------------------
__global__ __launch_bounds__(32 * WAVES) void node_kernel(
    const float* __restrict__ h, const bf16_t* __restrict__ hb,
    const float* __restrict__ agg,
    const float* __restrict__ nb1, const float* __restrict__ nb2,
    const bf16_t* __restrict__ nwb1, const bf16_t* __restrict__ nwb2,
    float* __restrict__ hout) {

    __shared__ __align__(16) bf16_t at[WAVES][16][2 * DD];

    const int lane = threadIdx.x & 31;
    const int wave = threadIdx.x >> 5;
    const int tile = blockIdx.x * WAVES + wave;
    if (tile >= N_NODES / 16) return;            // wave-uniform

    const long long n0 = (long long)tile * 16;

    // Stage [h_bf16 | bf16(agg)] tile.
    for (int idx = lane; idx < 256; idx += 32) {
        int m = idx >> 4, v = idx & 15;
        ((uint4*)&at[wave][m][0])[v] = ((const uint4*)(hb + (n0 + m) * DD))[v];
    }
    for (int idx = lane; idx < 2048; idx += 32) {
        int m = idx >> 7, k = idx & 127;
        at[wave][m][DD + k] = (bf16_t)agg[(n0 + m) * DD + k];
    }
    asm volatile("s_wait_dscnt 0" ::: "memory");

    const int half = lane >> 4;
    const int m16  = lane & 15;
    const bf16_t* aRow = &at[wave][m16][0];

    // ---- layer 1 (SiLU) ----
    v8f acc[8];
#pragma unroll
    for (int t = 0; t < 8; ++t) {
        float b = nb1[t * 16 + m16];
#pragma unroll
        for (int r = 0; r < 8; ++r) acc[t][r] = b;
    }
    tile_mlp<8>(aRow, (const v16bf*)nwb1, lane, half, acc);

#pragma unroll
    for (int t = 0; t < 8; ++t)
#pragma unroll
        for (int r = 0; r < 8; ++r)
            at[wave][r + 8 * half][t * 16 + m16] = (bf16_t)silu_f(acc[t][r]);
    asm volatile("s_wait_dscnt 0" ::: "memory");

    // ---- layer 2 (no activation) + residual ----
    v8f acc2[8];
#pragma unroll
    for (int t = 0; t < 8; ++t) {
        float b = nb2[t * 16 + m16];
#pragma unroll
        for (int r = 0; r < 8; ++r) acc2[t][r] = b;
    }
    tile_mlp<4>(aRow, (const v16bf*)nwb2, lane, half, acc2);

#pragma unroll
    for (int r = 0; r < 8; ++r) {
        int m = r + 8 * half;
        long long node = n0 + m;
#pragma unroll
        for (int t = 0; t < 8; ++t) {
            int n = t * 16 + m16;
            hout[node * DD + n] = h[node * DD + n] + acc2[t][r];
        }
    }
}

// ---------------------------------------------------------------------------
extern "C" void kernel_launch(void* const* d_in, const int* in_sizes, int n_in,
                              void* d_out, int out_size, void* d_ws, size_t ws_size,
                              hipStream_t stream) {
    const float*     h   = (const float*)d_in[0];
    const long long* ei  = (const long long*)d_in[1];   // int64 edge_index [2, E]
    const float*     ew1 = (const float*)d_in[2];
    const float*     eb1 = (const float*)d_in[3];
    const float*     ew2 = (const float*)d_in[4];
    const float*     eb2 = (const float*)d_in[5];
    const float*     nw1 = (const float*)d_in[6];
    const float*     nb1 = (const float*)d_in[7];
    const float*     nw2 = (const float*)d_in[8];
    const float*     nb2 = (const float*)d_in[9];

    // Workspace layout (all offsets 32B-aligned).
    char*   ws   = (char*)d_ws;
    bf16_t* hb   = (bf16_t*)(ws);                 // 12,800,000 B  bf16 h table
    float*  agg  = (float*) (ws + 12800000);      // 25,600,000 B  scatter-sum scratch
    bf16_t* wb1  = (bf16_t*)(ws + 38400000);      //     65,536 B
    bf16_t* wb2  = (bf16_t*)(ws + 38465536);      //     32,768 B
    bf16_t* nwb1 = (bf16_t*)(ws + 38498304);      //     65,536 B
    bf16_t* nwb2 = (bf16_t*)(ws + 38563840);      //     32,768 B

    float* hout = (float*)d_out;
    float* mij  = (float*)d_out + (size_t)N_NODES * DD;

    prep_weights<<<(8 * 8 * 32 * 16 + 255) / 256, 256, 0, stream>>>(
        ew1, ew2, nw1, nw2, wb1, wb2, nwb1, nwb2);
    conv_h<<<(N_NODES * DD + 255) / 256, 256, 0, stream>>>(h, hb, agg);

    edge_kernel<<<(N_EDGES / 16 + WAVES - 1) / WAVES, 32 * WAVES, 0, stream>>>(
        hb, ei, eb1, eb2, wb1, wb2, mij, agg);
    node_kernel<<<(N_NODES / 16 + WAVES - 1) / WAVES, 32 * WAVES, 0, stream>>>(
        h, hb, agg, nb1, nb2, nwb1, nwb2, hout);
}
